// LSTM_79748952752757
// MI455X (gfx1250) — compile-verified, matrix operand order checked by reading, not run on hardware
//
#include <hip/hip_runtime.h>
#include <hip/hip_bf16.h>

// Problem dims
#define SEQ   8192
#define EMB   100
#define Hd    75
#define G4    300      // 4*H gates per direction
#define KE    128      // padded K for embedding GEMM (100 -> 128, 4 x K32)
#define GP    608      // padded gate columns (2*300 -> 608, 38 x 16)
#define HIDP  160      // padded K for classifier GEMM (150 -> 160, 5 x K32)
#define NLAB  128

typedef __attribute__((ext_vector_type(16))) __bf16 v16bf;
typedef __attribute__((ext_vector_type(8)))  float  v8f;

// CDNA5 16-bit A/B fragment K index for half slot t (0..15) of a wave lane.
// ISA 7.12.2: VGPR j<4 -> K = 2j,2j+1 ; j>=4 -> K = 16+2(j-4)..; lanes 16-31 add 8.
// Per lane this is two contiguous 16-byte runs -> backend fuses to global_load_b128.
__device__ __forceinline__ int kmap(int t, int laneHi) {
  int j = t >> 1, h = t & 1;
  return ((j & 4) << 2) + (laneHi << 3) + ((j & 3) << 1) + h;
}

// ---------------- pack kernels (fp32 -> zero-padded bf16 operands) ----------
__global__ __launch_bounds__(256) void pack_embed(const int* __restrict__ x,
                                                  const float* __restrict__ emb,
                                                  __bf16* __restrict__ xsb) {
  int idx = blockIdx.x * 256 + threadIdx.x;        // SEQ*KE threads
  int t = idx / KE, k = idx % KE;
  float v = (k < EMB) ? emb[x[t] * EMB + k] : 0.0f;
  xsb[idx] = (__bf16)v;
}

__global__ __launch_bounds__(256) void pack_wih(const float* __restrict__ Wf,
                                                const float* __restrict__ Wb,
                                                __bf16* __restrict__ wp) {
  int idx = blockIdx.x * 256 + threadIdx.x;        // GP*KE threads
  int g = idx / KE, k = idx % KE;
  float v = 0.0f;
  if (k < EMB) {
    if (g < G4)            v = Wf[g * EMB + k];
    else if (g < 2 * G4)   v = Wb[(g - G4) * EMB + k];
  }
  wp[idx] = (__bf16)v;
}

__global__ __launch_bounds__(256) void pack_wlin(const float* __restrict__ W,
                                                 __bf16* __restrict__ wp) {
  int idx = blockIdx.x * 256 + threadIdx.x;        // NLAB*HIDP threads
  int l = idx / HIDP, k = idx % HIDP;
  float v = (k < 2 * Hd) ? W[l * (2 * Hd) + k] : 0.0f;
  wp[idx] = (__bf16)v;
}

__global__ __launch_bounds__(256) void pad_hpack(__bf16* __restrict__ hp) {
  int idx = blockIdx.x * 256 + threadIdx.x;        // SEQ*10 threads
  int t = idx / 10, k = 2 * Hd + idx % 10;         // cols 150..159
  hp[t * HIDP + k] = (__bf16)0.0f;
}

// ---------------- generic bf16 WMMA GEMM: C[M,N] = A[M,K] * B[N,K]^T --------
// One 16x16 output tile per wave; KT compile-time -> fully unrolled wmma chain.
template <int KT, bool HAS_BIAS>
__global__ __launch_bounds__(256)
void gemm_bf16_wmma(const __bf16* __restrict__ A, int lda,
                    const __bf16* __restrict__ B, int ldb,
                    float* __restrict__ C, int ldc,
                    const float* __restrict__ bias,
                    int ntiles) {
  const int wave   = threadIdx.x >> 5;
  const int lane   = threadIdx.x & 31;
  const int laneHi = (lane >> 4) & 1;
  const int tile   = blockIdx.x * 8 + wave;
  const int mtile  = tile / ntiles;
  const int ntile  = tile % ntiles;

  const __bf16* arow = A + (size_t)(mtile * 16 + (lane & 15)) * lda;
  const __bf16* brow = B + (size_t)(ntile * 16 + (lane & 15)) * ldb;

  v8f acc = {};
#pragma unroll
  for (int kk = 0; kk < KT; ++kk) {
    union { v16bf v; __bf16 e[16]; } af, bf_;
    const int kb = kk * 32;
#pragma unroll
    for (int t = 0; t < 16; ++t) {
      const int k = kb + kmap(t, laneHi);
      af.e[t]  = arow[k];
      bf_.e[t] = brow[k];
    }
    acc = __builtin_amdgcn_wmma_f32_16x16x32_bf16(false, af.v, false, bf_.v,
                                                  (short)0, acc, false, false);
  }

  const int n  = lane & 15;
  const int gn = ntile * 16 + n;
  const float bv = HAS_BIAS ? bias[gn] : 0.0f;
  float* crow = C + (size_t)(mtile * 16 + (laneHi << 3)) * ldc + gn;
#pragma unroll
  for (int v = 0; v < 8; ++v) {
    crow[(size_t)v * ldc] = acc[v] + bv;
  }
}

// ---------------- serial LSTM scan, one workgroup per direction -------------
// 320 threads: j<300 each own one gate row of W_hh (75 regs); j<75 do the
// pointwise cell update. h lives in LDS (broadcast reads), c in registers.
__global__ __launch_bounds__(320)
void lstm_scan(const float* __restrict__ pre,            // [SEQ][GP]
               const float* __restrict__ W_hh_f, const float* __restrict__ W_hh_b,
               const float* __restrict__ b_ih_f, const float* __restrict__ b_hh_f,
               const float* __restrict__ b_ih_b, const float* __restrict__ b_hh_b,
               const float* __restrict__ h0, const float* __restrict__ c0,
               __bf16* __restrict__ hpack) {               // [SEQ][HIDP]
  const int dir = blockIdx.x;                              // 0 = fwd, 1 = bwd
  const int j   = threadIdx.x;
  const float* Whh = dir ? W_hh_b : W_hh_f;
  const float* bi  = dir ? b_ih_b : b_ih_f;
  const float* bh  = dir ? b_hh_b : b_hh_f;

  __shared__ float h_s[Hd];
  __shared__ float g_s[G4];

  float w[Hd];
  float bj = 0.0f;
  if (j < G4) {
    bj = bi[j] + bh[j];
#pragma unroll
    for (int k = 0; k < Hd; ++k) w[k] = Whh[j * Hd + k];
  }
  float c = 0.0f;
  if (j < Hd) {
    h_s[j] = h0[dir * Hd + j];
    c      = c0[dir * Hd + j];
  }
  __syncthreads();

  for (int s = 0; s < SEQ; ++s) {
    const int t = dir ? (SEQ - 1 - s) : s;
    if (j < G4) {
      if (s + 1 < SEQ) {
        const int tn = dir ? (SEQ - 2 - s) : (s + 1);
        __builtin_prefetch(&pre[(size_t)tn * GP + dir * G4 + j], 0, 3);
      }
      float g = pre[(size_t)t * GP + dir * G4 + j] + bj;
#pragma unroll
      for (int k = 0; k < Hd; ++k) g = __builtin_fmaf(w[k], h_s[k], g);
      g_s[j] = g;
    }
    __syncthreads();
    if (j < Hd) {
      const float ig = 1.0f / (1.0f + __expf(-g_s[j]));
      const float fg = 1.0f / (1.0f + __expf(-g_s[Hd + j]));
      const float gg = tanhf(g_s[2 * Hd + j]);
      const float og = 1.0f / (1.0f + __expf(-g_s[3 * Hd + j]));
      c = fg * c + ig * gg;
      const float hn = og * tanhf(c);
      h_s[j] = hn;
      hpack[(size_t)t * HIDP + dir * Hd + j] = (__bf16)hn;
    }
    __syncthreads();
  }
}

// ---------------- launcher ---------------------------------------------------
extern "C" void kernel_launch(void* const* d_in, const int* in_sizes, int n_in,
                              void* d_out, int out_size, void* d_ws, size_t ws_size,
                              hipStream_t stream) {
  (void)in_sizes; (void)n_in; (void)out_size; (void)ws_size;
  const int*   x      = (const int*)  d_in[0];
  const float* emb    = (const float*)d_in[1];
  const float* W_ih_f = (const float*)d_in[2];
  const float* W_hh_f = (const float*)d_in[3];
  const float* b_ih_f = (const float*)d_in[4];
  const float* b_hh_f = (const float*)d_in[5];
  const float* W_ih_b = (const float*)d_in[6];
  const float* W_hh_b = (const float*)d_in[7];
  const float* b_ih_b = (const float*)d_in[8];
  const float* b_hh_b = (const float*)d_in[9];
  const float* W_lin  = (const float*)d_in[10];
  const float* b_lin  = (const float*)d_in[11];
  const float* h0     = (const float*)d_in[12];
  const float* c0     = (const float*)d_in[13];
  float* out = (float*)d_out;

  char* ws = (char*)d_ws;
  __bf16* xsb   = (__bf16*)(ws);                 // SEQ*KE      = 2,097,152 B
  __bf16* wihp  = (__bf16*)(ws + 2097152);       // GP*KE       =   155,648 B
  __bf16* wlinp = (__bf16*)(ws + 2252800);       // NLAB*HIDP   =    40,960 B
  __bf16* hpack = (__bf16*)(ws + 2293760);       // SEQ*HIDP    = 2,621,440 B
  float*  pre   = (float*) (ws + 4915200);       // SEQ*GP*4    = 19,922,944 B

  // 1) pack operands to bf16 (zero-padded)
  pack_embed<<<(SEQ * KE) / 256, 256, 0, stream>>>(x, emb, xsb);
  pack_wih  <<<(GP * KE) / 256, 256, 0, stream>>>(W_ih_f, W_ih_b, wihp);
  pack_wlin <<<(NLAB * HIDP) / 256, 256, 0, stream>>>(W_lin, wlinp);
  pad_hpack <<<(SEQ * 10) / 256, 256, 0, stream>>>(hpack);

  // 2) pre[t][g] = xs @ [W_ih_f;W_ih_b]^T  (512 x 38 tiles, 8 waves/block)
  gemm_bf16_wmma<KE / 32, false><<<(SEQ / 16) * (GP / 16) / 8, 256, 0, stream>>>(
      xsb, KE, wihp, KE, pre, GP, nullptr, GP / 16);

  // 3) the two serial scans, one WGP each
  lstm_scan<<<2, 320, 0, stream>>>(pre, W_hh_f, W_hh_b, b_ih_f, b_hh_f,
                                   b_ih_b, b_hh_b, h0, c0, hpack);

  // 4) out[t][l] = concat(h_f,h_b) @ W_lin^T + b_lin  (512 x 8 tiles)
  gemm_bf16_wmma<HIDP / 32, true><<<(SEQ / 16) * (NLAB / 16) / 8, 256, 0, stream>>>(
      hpack, HIDP, wlinp, HIDP, out, NLAB, b_lin, NLAB / 16);
}
